// ATT_learner_30227979829651
// MI455X (gfx1250) — compile-verified, matrix operand order checked by reading, not run on hardware
//
#include <hip/hip_runtime.h>

// ---------------------------------------------------------------------------
// ATT graph-learner: emb = normalize(relu(x*w0)*w1); sim = emb@emb^T;
// out = relu(sim * topk_mask(sim, k+1)).  N=10000, D=128, k=30.
//
// MI455X strategy:
//  * GEMM is 25.6 GFLOPs (trivial); HBM floor is the 400 MB output write
//    (~17us @ 23.3 TB/s). Never materialize sim in HBM: fuse compute with an
//    exact streaming top-31 per row; only zeros + 31 entries/row reach HBM.
//  * Operands in one WMMA-layout-packed bf16 buffer (~2.6 MB, L2-resident);
//    A (16x32, M-striped) and B (32x16, N-striped) layouts coincide for
//    E·E^T, so one packed image serves both. 32B/lane b128 operand loads.
//  * 64 rows/block (4 A row-tiles in VGPRs): each B tile load feeds 16 WMMAs
//    on 4 independent accumulator chains -> L2 B-traffic drops to ~430 MB
//    (below the HBM output floor) and load latency amortizes 16x.
// ---------------------------------------------------------------------------

#define DIM 128
#define ROWS 64                      // rows per block (4 row-tiles)
#define ROWTILES (ROWS / 16)
#define CHUNK 160                    // columns per LDS staging chunk
#define TILES_PER_CHUNK (CHUNK / 16) // 10
#define TOPK 31                      // k+1
#define NEG_INF (-3.0e38f)
#define POS_INF (3.0e38f)

typedef __bf16 bf16_t;
typedef bf16_t v16bf __attribute__((ext_vector_type(16)));
typedef float  v8f   __attribute__((ext_vector_type(8)));

// pack element index: tile (16 rows/cols of emb), kstep (32 K), lane, element
__device__ __forceinline__ size_t pack_index(int tile, int ks, int lane, int e) {
    return ((((size_t)tile * 4 + ks) * 32) + lane) * 16 + e;
}

// ---------------------------------------------------------------------------
// Kernel 0a: clear the N*N output (the result is mostly zeros).
// ---------------------------------------------------------------------------
__global__ __launch_bounds__(256) void zero_out_kernel(float4* __restrict__ out4,
                                                       size_t n4) {
    size_t i = (size_t)blockIdx.x * blockDim.x + threadIdx.x;
    size_t stride = (size_t)gridDim.x * blockDim.x;
    float4 z = {0.f, 0.f, 0.f, 0.f};
    for (; i < n4; i += stride) out4[i] = z;
}

// Kernel 0b: clear the packed operand buffer (covers padded tail tiles).
__global__ __launch_bounds__(256) void zero_pack_kernel(float4* __restrict__ p4,
                                                        size_t n4) {
    size_t i = (size_t)blockIdx.x * blockDim.x + threadIdx.x;
    size_t stride = (size_t)gridDim.x * blockDim.x;
    float4 z = {0.f, 0.f, 0.f, 0.f};
    for (; i < n4; i += stride) p4[i] = z;
}

// ---------------------------------------------------------------------------
// Kernel 1: h = relu(x*w0)*w1; emb = h/max(||h||,eps); emit bf16 directly in
// the WMMA tile-register image:
//   pack[tile][ks][lane][e] = emb[tile*16 + (lane&15)]
//                                [ks*32 + 8*(lane>>4) + ((e&7) + (e>=8?16:0))]
// One wave per row; each lane owns 4 of the 128 features.
// ---------------------------------------------------------------------------
__global__ __launch_bounds__(256) void embed_kernel(const float* __restrict__ x,
                                                    const float* __restrict__ w0,
                                                    const float* __restrict__ w1,
                                                    bf16_t* __restrict__ pack,
                                                    int N) {
    int wave = threadIdx.x >> 5;
    int lane = threadIdx.x & 31;
    int row  = blockIdx.x * 8 + wave;
    if (row >= N) return;

    int base = lane * 4;
    float h[4];
    float s = 0.f;
#pragma unroll
    for (int j = 0; j < 4; ++j) {
        float v = x[(size_t)row * DIM + base + j];
        float t = v * w0[base + j];
        t = t > 0.f ? t : 0.f;
        t *= w1[base + j];
        h[j] = t;
        s += t * t;
    }
#pragma unroll
    for (int off = 16; off; off >>= 1) s += __shfl_xor(s, off, 32);
    float inv = 1.0f / fmaxf(sqrtf(s), 1e-12f);

    const int tile = row >> 4;
    const int m    = row & 15;
#pragma unroll
    for (int j = 0; j < 4; ++j) {
        int k  = base + j;
        int ks = k >> 5;
        int kl = k & 31;
        int half   = (kl >> 3) & 1;
        int lane_t = m + 16 * half;
        int e      = (kl & 7) + (kl >= 16 ? 8 : 0);
        pack[pack_index(tile, ks, lane_t, e)] = (bf16_t)(h[j] * inv);
    }
}

// ---------------------------------------------------------------------------
// Kernel 2: fused sim + exact top-31 + scatter.
// Block = 64 output rows, 512 threads = 16 waves.
//   compute : waves 0..9 each own one 16-col tile of the 160-col chunk; each
//             loads 4 B ksteps up front (8x b128, one clause) then fires 16
//             v_wmma_f32_16x16x32_bf16 on 4 independent row-tile acc chains.
//   select  : wave w owns rows {w, w+16, w+32, w+48}; wave-argmax + min-slot
//             replacement into 31-entry running lists, threshold early-exit.
//   write   : relu(top values) scattered to out.
// ---------------------------------------------------------------------------
__global__ __launch_bounds__(512) void sim_topk_kernel(const v16bf* __restrict__ packv,
                                                       float* __restrict__ out,
                                                       int N, int tilesP) {
    __shared__ float simTile[ROWS * CHUNK];      // 40 KB
    __shared__ float topv[ROWS][32];             // 8 KB
    __shared__ int   topi[ROWS][32];             // 8 KB

    const int tid  = threadIdx.x;
    const int wave = tid >> 5;
    const int lane = tid & 31;
    const int half = lane >> 4;
    const int l16  = lane & 15;
    const int rowBase = blockIdx.x * ROWS;
    const int rowTile0 = blockIdx.x * ROWTILES;

    // A tiles: 4 row-tiles x 4 ksteps, one 32B packed slice per lane (128 VGPRs)
    v16bf A[ROWTILES][4];
#pragma unroll
    for (int rt = 0; rt < ROWTILES; ++rt)
#pragma unroll
        for (int ks = 0; ks < 4; ++ks)
            A[rt][ks] = packv[((size_t)(rowTile0 + rt) * 4 + ks) * 32 + lane];

    // top lists: 31 live slots at -inf, slot 31 = +inf sentinel
    for (int idx = tid; idx < ROWS * 32; idx += 512) {
        int r = idx >> 5, l = idx & 31;
        topv[r][l] = (l < TOPK) ? NEG_INF : POS_INF;
        topi[r][l] = 0;
    }
    __syncthreads();

    const int numChunks = (N + CHUNK - 1) / CHUNK;
    for (int c = 0; c < numChunks; ++c) {
        const int cb = c * CHUNK;

        // ---------------- compute: 10 col tiles, waves 0..9 ----------------
        for (int t = wave; t < TILES_PER_CHUNK; t += 16) {
            const int gtile = c * TILES_PER_CHUNK + t;

            v16bf B[4];
#pragma unroll
            for (int ks = 0; ks < 4; ++ks)
                B[ks] = packv[((size_t)gtile * 4 + ks) * 32 + lane];

            v8f acc[ROWTILES];
#pragma unroll
            for (int rt = 0; rt < ROWTILES; ++rt) {
                acc[rt] = (v8f){0.f, 0.f, 0.f, 0.f, 0.f, 0.f, 0.f, 0.f};
#pragma unroll
                for (int ks = 0; ks < 4; ++ks)
                    acc[rt] = __builtin_amdgcn_wmma_f32_16x16x32_bf16(
                        false, A[rt][ks], false, B[ks], (short)0, acc[rt],
                        false, false);
            }

            // pull next chunk's operand tile toward the WGP while we compute
            int ntile = gtile + TILES_PER_CHUNK;
            if (ntile < tilesP)
                __builtin_prefetch(&packv[(size_t)ntile * 4 * 32], 0, 1);

#pragma unroll
            for (int rt = 0; rt < ROWTILES; ++rt)
#pragma unroll
                for (int r = 0; r < 8; ++r)
                    simTile[(rt * 16 + r + 8 * half) * CHUNK + t * 16 + l16] =
                        acc[rt][r];
        }
        __syncthreads();

        // padded columns (beyond N) were computed from zeros -> force -inf
        if (cb + CHUNK > N) {
            for (int idx = tid; idx < ROWS * CHUNK; idx += 512) {
                int col = cb + (idx % CHUNK);
                if (col >= N) simTile[idx] = NEG_INF;
            }
            __syncthreads();
        }

        // ---------------- select: wave w owns rows w + 16q ----------------
#pragma unroll
        for (int q = 0; q < ROWS / 16; ++q) {
            const int row = wave + 16 * q;
            float* srow = &simTile[row * CHUNK];
            for (int iter = 0; iter < CHUNK; ++iter) {
                float lv = NEG_INF;
                int li = lane;
#pragma unroll
                for (int j = 0; j < CHUNK / 32; ++j) {
                    float v = srow[lane + 32 * j];
                    if (v > lv) { lv = v; li = lane + 32 * j; }
                }
#pragma unroll
                for (int off = 16; off; off >>= 1) {       // wave argmax
                    float ov = __shfl_xor(lv, off, 32);
                    int   oi = __shfl_xor(li, off, 32);
                    if (ov > lv || (ov == lv && oi < li)) { lv = ov; li = oi; }
                }
                float sv = topv[row][lane];
                int   ss = lane;
#pragma unroll
                for (int off = 16; off; off >>= 1) {       // argmin slot
                    float ov = __shfl_xor(sv, off, 32);
                    int   oi = __shfl_xor(ss, off, 32);
                    if (ov < sv || (ov == sv && oi < ss)) { sv = ov; ss = oi; }
                }
                if (lv <= sv) break;   // chunk max can't improve top-31
                if (lane == ss) { topv[row][ss] = lv; topi[row][ss] = cb + li; }
                if (lane == (li & 31)) srow[li] = NEG_INF;
            }
        }
        __syncthreads();
    }

    // ---------------- scatter: relu(top-31) per row ----------------
#pragma unroll
    for (int q = 0; q < ROWS / 16; ++q) {
        const int row = rowBase + wave + 16 * q;
        if (row < N && lane < TOPK) {
            float v  = topv[wave + 16 * q][lane];
            int col  = topi[wave + 16 * q][lane];
            if (v > 0.5f * NEG_INF)
                out[(size_t)row * N + col] = fmaxf(v, 0.0f);
        }
    }
}

// ---------------------------------------------------------------------------
// Host-side launch. Inputs: features[N*128] f32, w0[128] f32, w1[128] f32, k.
// Output: N*N f32. Workspace: pack = tilesP x 4 x 32 x 16 bf16 (~2.6 MB).
// ---------------------------------------------------------------------------
extern "C" void kernel_launch(void* const* d_in, const int* in_sizes, int n_in,
                              void* d_out, int out_size, void* d_ws, size_t ws_size,
                              hipStream_t stream) {
    const float* feat = (const float*)d_in[0];
    const float* w0   = (const float*)d_in[1];
    const float* w1   = (const float*)d_in[2];
    float* out        = (float*)d_out;

    const int N  = in_sizes[0] / DIM;                    // 10000
    const int numChunks = (N + CHUNK - 1) / CHUNK;       // 63
    const int tilesP    = numChunks * TILES_PER_CHUNK;   // 630 col tiles

    bf16_t* pack = (bf16_t*)d_ws;                        // [tilesP][4][32][16] bf16
    const size_t packBytes = (size_t)tilesP * 4 * 32 * 16 * sizeof(bf16_t);

    // 1) zero the (mostly-zero) output and the packed operand buffer
    size_t total = (size_t)N * (size_t)N;
    zero_out_kernel<<<4096, 256, 0, stream>>>((float4*)out, total / 4);
    zero_pack_kernel<<<256, 256, 0, stream>>>((float4*)pack, packBytes / 16);

    // 2) embed + normalize -> bf16 in WMMA tile-register image
    embed_kernel<<<(N + 7) / 8, 256, 0, stream>>>(feat, w0, w1, pack, N);

    // 3) fused WMMA sim + streaming exact top-31 + relu scatter
    sim_topk_kernel<<<(N + ROWS - 1) / ROWS, 512, 0, stream>>>(
        (const v16bf*)pack, out, N, tilesP);
}
